// VectorQuantizer_86517821215082
// MI455X (gfx1250) — matrix operand order repaired and checked by reading, not compiled, not run on hardware
//
#include <hip/hip_runtime.h>

// ---------------------------------------------------------------------------
// VQ codebook lookup: z [16,256,32,32] f32, codebook [16384,256] f32
// N = 16384 rows, M = 16384 codes, K = 256.  GEMM (137 GFLOP) + row argmin.
// bf16 WMMA w/ f32 accum (error << top1/top2 gap since |e| <= 1/16384),
// exact f32 ||e||^2 added, exact f32 gather for z_q.
// ---------------------------------------------------------------------------
#define NROWS 16384
#define NCODES 16384
#define DIM 256
#define DIMU (DIM / 2)      // row length in packed uint32 (2 bf16 each) = 128
#define TILE_M 64           // codebook rows staged in LDS per step
#define M_SLICES 8          // M split across blocks, merged via atomicMin(u64)
#define M_PER_SLICE (NCODES / M_SLICES) // 2048
#define NTILES (M_PER_SLICE / TILE_M)   // 32

// Workspace layout (bytes)
#define WS_ZBF_OFF   0u                      // 8 MB bf16 z [N][K]
#define WS_CBBF_OFF  8388608u                // 8 MB bf16 codebook [M][K]
#define WS_NORMS_OFF 16777216u               // 64 KB f32 ||e||^2
#define WS_KEYS_OFF  16842752u               // 128 KB u64 (encoded dist | idx)

typedef __attribute__((ext_vector_type(16))) __bf16 v16bf;
typedef __attribute__((ext_vector_type(8)))  float  v8f;
typedef __attribute__((ext_vector_type(8)))  unsigned int v8u;

union FragU { v8u u; v16bf b; };

__device__ __forceinline__ unsigned short f32_to_bf16(float f) {
  unsigned int u = __float_as_uint(f);
  unsigned int rnd = 0x7FFFu + ((u >> 16) & 1u);   // round-to-nearest-even
  return (unsigned short)((u + rnd) >> 16);
}

// monotone encoding of float for unsigned min: order-preserving
__device__ __forceinline__ unsigned int enc_f32(float f) {
  unsigned int b = __float_as_uint(f);
  return (b & 0x80000000u) ? ~b : (b | 0x80000000u);
}

__device__ __forceinline__ void wait_asynccnt0() {
#if __has_builtin(__builtin_amdgcn_s_wait_asynccnt)
  __builtin_amdgcn_s_wait_asynccnt(0);
#else
  asm volatile("s_wait_asynccnt 0x0" ::: "memory");
#endif
}

// Pin issue pattern: N DS-reads then M WMMAs (per K-step software pipeline).
// sched_group_barrier args must be integer constants -> template params.
template <int N>
__device__ __forceinline__ void sched_ds() {
#if __has_builtin(__builtin_amdgcn_sched_group_barrier)
  __builtin_amdgcn_sched_group_barrier(0x100, N, 0);   // DS read group
#endif
}
template <int N>
__device__ __forceinline__ void sched_wmma() {
#if __has_builtin(__builtin_amdgcn_sched_group_barrier)
  __builtin_amdgcn_sched_group_barrier(0x008, N, 0);   // MFMA/WMMA group
#endif
}

// ---------------------------------------------------------------------------
// Kernel 1: z [B,C,H,W] f32 -> z_bf [n = b*1024+hw][c] bf16 (read-coalesced)
// ---------------------------------------------------------------------------
__global__ __launch_bounds__(256) void vq_convert_z(
    const float* __restrict__ z, unsigned short* __restrict__ z_bf) {
  unsigned int gid = blockIdx.x * 256u + threadIdx.x;   // 4,194,304 total
  unsigned int hw = gid & 1023u;
  unsigned int c  = (gid >> 10) & 255u;
  unsigned int b  = gid >> 18;
  float v = z[gid];
  unsigned int n = (b << 10) + hw;
  z_bf[(size_t)n * DIM + c] = f32_to_bf16(v);
}

// ---------------------------------------------------------------------------
// Kernel 2: codebook -> bf16 + exact f32 norms  (one row per 256-thread block)
// ---------------------------------------------------------------------------
__global__ __launch_bounds__(256) void vq_prep_cb(
    const float* __restrict__ cb, unsigned short* __restrict__ cb_bf,
    float* __restrict__ norms) {
  int m = blockIdx.x;
  int t = threadIdx.x;
  float v = cb[(size_t)m * DIM + t];
  cb_bf[(size_t)m * DIM + t] = f32_to_bf16(v);
  float s = v * v;
  #pragma unroll
  for (int off = 16; off > 0; off >>= 1) s += __shfl_down(s, off, 32);
  __shared__ float acc[8];
  if ((t & 31) == 0) acc[t >> 5] = s;
  __syncthreads();
  if (t == 0) {
    float tot = 0.f;
    #pragma unroll
    for (int i = 0; i < 8; ++i) tot += acc[i];
    norms[m] = tot;
  }
}

// ---------------------------------------------------------------------------
// Kernel 3: init per-row argmin keys to +inf
// ---------------------------------------------------------------------------
__global__ __launch_bounds__(256) void vq_init_keys(unsigned long long* __restrict__ keys) {
  unsigned int gid = blockIdx.x * 256u + threadIdx.x;
  keys[gid] = 0xFFFFFFFFFFFFFFFFull;
}

// ---------------------------------------------------------------------------
// Kernel 4: WMMA GEMM (scores = z . cb^T) fused with argmin of
//           d = ||e||^2 - 2*score, merged via atomicMin on packed u64.
// grid = (NROWS/128, M_SLICES); block = 256 (8 waves, 16 rows each).
// Double-buffered LDS tiles filled by GLOBAL_LOAD_ASYNC_TO_LDS_B128
// (ASYNCcnt) + software-pipelined B fragments whose (DS8, WMMA4) issue
// pattern is enforced with sched_group_barrier so LDS latency overlaps
// the matrix pipe.
// ---------------------------------------------------------------------------
__global__ __launch_bounds__(256) void vq_gemm_argmin(
    const unsigned short* __restrict__ z_bf,
    const unsigned short* __restrict__ cb_bf,
    const float* __restrict__ norms,
    unsigned long long* __restrict__ keys) {
  __shared__ __align__(16) unsigned int ldsB[2][TILE_M * DIMU];  // 2 x 32 KB

  const int tid  = threadIdx.x;
  const int wave = tid >> 5;
  const int lane = tid & 31;
  const int li   = lane & 15;   // row (A) / column (B/D) within tile
  const int h    = lane >> 4;   // lane-half selects K/M sub-block

  const int rowBase = blockIdx.x * 128 + wave * 16;
  const int mBase   = blockIdx.y * M_PER_SLICE;

  // Per-thread async-copy slice: 128 contiguous bytes of one codebook row.
  const int sr   = tid >> 2;            // 0..63 : row within tile
  const int sseg = (tid & 3) * 32;      // uint offset within 128-uint row

  // Async stage: copy 64 bf16 codebook rows into ldsB[buf].
  // ISA: the instruction offset is added to both the LDS and the global address,
  // so one LDS-address VGPR + one global pointer covers all 8 B128s.
  auto stage = [&](int buf, int m0) {
    unsigned lds_off = (unsigned)(unsigned long long)&ldsB[buf][sr * DIMU + sseg];
    const unsigned int* gsrc =
        (const unsigned int*)cb_bf + (size_t)(m0 + sr) * DIMU + sseg;
    asm volatile(
        "global_load_async_to_lds_b128 %0, %1, off\n\t"
        "global_load_async_to_lds_b128 %0, %1, off offset:16\n\t"
        "global_load_async_to_lds_b128 %0, %1, off offset:32\n\t"
        "global_load_async_to_lds_b128 %0, %1, off offset:48\n\t"
        "global_load_async_to_lds_b128 %0, %1, off offset:64\n\t"
        "global_load_async_to_lds_b128 %0, %1, off offset:80\n\t"
        "global_load_async_to_lds_b128 %0, %1, off offset:96\n\t"
        "global_load_async_to_lds_b128 %0, %1, off offset:112"
        :: "v"(lds_off), "v"(gsrc) : "memory");
  };

  // B fragment load from LDS (2 x ds_load_b128 into a dedicated register set)
  // B layout: lane L: col = L%16; elem e: K = kt*32 + (e/8)*16 + (L/16)*8 + e%8
  auto loadB = [&](FragU& dst, int buf, int cs, int kt) {
    const unsigned int* bp =
        &ldsB[buf][(cs * 16 + li) * DIMU + kt * 16 + h * 4];
    uint4 b0 = ((const uint4*)bp)[0];
    uint4 b1 = ((const uint4*)bp)[2];
    dst.u[0] = b0.x; dst.u[1] = b0.y; dst.u[2] = b0.z; dst.u[3] = b0.w;
    dst.u[4] = b1.x; dst.u[5] = b1.y; dst.u[6] = b1.z; dst.u[7] = b1.w;
  };

  // --- Load this wave's A tile (16 rows x 256 K) as 8 bf16 fragments -------
  // A layout: lane L: row = L%16; elem e: K = (e/8)*16 + (L/16)*8 + e%8
  FragU afrag[8];
  {
    const unsigned int* zr =
        (const unsigned int*)z_bf + (size_t)(rowBase + li) * DIMU;
    #pragma unroll
    for (int kt = 0; kt < 8; ++kt) {
      uint4 a0 = *(const uint4*)(zr + kt * 16 + h * 4);       // K = kt*32 + h*8 ..
      uint4 a1 = *(const uint4*)(zr + kt * 16 + 8 + h * 4);   // K = kt*32+16+h*8 ..
      afrag[kt].u[0] = a0.x; afrag[kt].u[1] = a0.y;
      afrag[kt].u[2] = a0.z; afrag[kt].u[3] = a0.w;
      afrag[kt].u[4] = a1.x; afrag[kt].u[5] = a1.y;
      afrag[kt].u[6] = a1.z; afrag[kt].u[7] = a1.w;
    }
  }

  float best[8];
  int   bidx[8];
  #pragma unroll
  for (int v = 0; v < 8; ++v) { best[v] = 3.4e38f; bidx[v] = 0; }

  stage(0, mBase);                       // prologue: fill buffer 0
  int cur = 0;

  for (int it = 0; it < NTILES; ++it) {
    const int m0 = mBase + it * TILE_M;
    wait_asynccnt0();                    // this wave's async copy done
    __syncthreads();                     // all waves' copies done / buf free
    if (it + 1 < NTILES) stage(cur ^ 1, m0 + TILE_M);  // overlap next copy

    // --- 4 independent accumulators, K-steps software-pipelined -----------
    v8f acc[4];
    #pragma unroll
    for (int cs = 0; cs < 4; ++cs) acc[cs] = (v8f){0,0,0,0,0,0,0,0};

    FragU bfr[2][4];                     // ping-pong B fragment sets
    #pragma unroll
    for (int cs = 0; cs < 4; ++cs) loadB(bfr[0][cs], cur, cs, 0);
    sched_ds<8>();                       // prologue fragment loads first

    #pragma unroll
    for (int kt = 0; kt < 8; ++kt) {
      const int pc = kt & 1;             // current set
      const int pn = pc ^ 1;             // prefetch set
      if (kt + 1 < 8) {
        #pragma unroll
        for (int cs = 0; cs < 4; ++cs) loadB(bfr[pn][cs], cur, cs, kt + 1);
      }
      #pragma unroll
      for (int cs = 0; cs < 4; ++cs) {
        acc[cs] = __builtin_amdgcn_wmma_f32_16x16x32_bf16(
            false, afrag[kt].b, false, bfr[pc][cs].b, (short)0, acc[cs],
            false, false);
      }
      // Enforce issue pattern: 8 prefetch DS-reads, then 4 WMMAs.
      if (kt + 1 < 8) sched_ds<8>();
      sched_wmma<4>();
    }

    // --- Fused argmin update: d = ||e||^2 - 2*score ------------------------
    #pragma unroll
    for (int cs = 0; cs < 4; ++cs) {
      const int col = m0 + cs * 16 + li;     // D: lane holds column li
      const float nrm = norms[col];
      #pragma unroll
      for (int v = 0; v < 8; ++v) {          // D elem v -> row v + 8*h
        float d = nrm - 2.0f * acc[cs][v];
        if (d < best[v]) { best[v] = d; bidx[v] = col; }
      }
    }
    cur ^= 1;
  }

  // --- Cross-lane reduce (16 column-lanes per row), then global merge ------
  __syncthreads();                        // done reading LDS; alias as scratch
  float* redVal = (float*)&ldsB[0][0];                    // [8][16][16] f32
  int*   redIdx = (int*)(&ldsB[0][0] + 2048);             // [8][16][16] i32
  #pragma unroll
  for (int v = 0; v < 8; ++v) {
    int wr = v + h * 8;
    redVal[(wave * 16 + wr) * 16 + li] = best[v];
    redIdx[(wave * 16 + wr) * 16 + li] = bidx[v];
  }
  __syncthreads();
  if (lane < 16) {
    float bv = redVal[(wave * 16 + lane) * 16 + 0];
    int   bi = redIdx[(wave * 16 + lane) * 16 + 0];
    #pragma unroll
    for (int s = 1; s < 16; ++s) {
      float v2 = redVal[(wave * 16 + lane) * 16 + s];
      int   i2 = redIdx[(wave * 16 + lane) * 16 + s];
      if (v2 < bv || (v2 == bv && i2 < bi)) { bv = v2; bi = i2; }
    }
    unsigned long long key =
        ((unsigned long long)enc_f32(bv) << 32) | (unsigned int)bi;
    atomicMin(&keys[rowBase + lane], key);   // tie -> lowest idx (matches jnp)
  }
}

// ---------------------------------------------------------------------------
// Kernel 5: gather z_q rows (exact f32 codebook) + emit idx as float
// ---------------------------------------------------------------------------
__global__ __launch_bounds__(64) void vq_gather(
    const float* __restrict__ cb, const unsigned long long* __restrict__ keys,
    float* __restrict__ out_zq, float* __restrict__ out_idx) {
  int n = blockIdx.x;
  int t = threadIdx.x;                       // 64 threads x float4 = 256
  int id = (int)(unsigned int)(keys[n] & 0xFFFFFFFFull);
  const float4* src = (const float4*)(cb + (size_t)id * DIM);
  float4* dst = (float4*)(out_zq + (size_t)n * DIM);
  dst[t] = src[t];
  if (t == 0) out_idx[n] = (float)id;
}

// ---------------------------------------------------------------------------
extern "C" void kernel_launch(void* const* d_in, const int* in_sizes, int n_in,
                              void* d_out, int out_size, void* d_ws, size_t ws_size,
                              hipStream_t stream) {
  const float* z  = (const float*)d_in[0];   // [16,256,32,32] f32
  const float* cb = (const float*)d_in[1];   // [16384,256] f32
  float* out = (float*)d_out;                // z_q (16384*256) then idx (16384)

  char* ws = (char*)d_ws;
  unsigned short* z_bf  = (unsigned short*)(ws + WS_ZBF_OFF);
  unsigned short* cb_bf = (unsigned short*)(ws + WS_CBBF_OFF);
  float* norms          = (float*)(ws + WS_NORMS_OFF);
  unsigned long long* keys = (unsigned long long*)(ws + WS_KEYS_OFF);

  vq_convert_z<<<(NROWS * DIM) / 256, 256, 0, stream>>>(z, z_bf);
  vq_prep_cb<<<NCODES, 256, 0, stream>>>(cb, cb_bf, norms);
  vq_init_keys<<<NROWS / 256, 256, 0, stream>>>(keys);

  dim3 grid(NROWS / 128, M_SLICES);
  vq_gemm_argmin<<<grid, 256, 0, stream>>>(z_bf, cb_bf, norms, keys);

  vq_gather<<<NROWS, 64, 0, stream>>>(cb, keys, out, out + (size_t)NROWS * DIM);
}
